// LatentDiffusionRebuilder_43267500540063
// MI455X (gfx1250) — compile-verified, matrix operand order checked by reading, not run on hardware
//
#include <hip/hip_runtime.h>
#include <hip/hip_bf16.h>

// ---------------- problem constants ----------------
#define D_      256
#define K_      4
#define B_      2048
#define L_      50
#define HID_    1024
#define TDIM_   128
#define IN_DIM_ 1152              // 4*D + TDIM
#define NSTEPS_ 10
#define MROWS_  (B_ * K_)         // 8192

typedef _Float16 h16;
typedef __attribute__((ext_vector_type(16))) _Float16 v16h;
typedef __attribute__((ext_vector_type(8)))  _Float16 v8h;
typedef __attribute__((ext_vector_type(8)))  float    v8f;

// =====================================================================
// Small utility kernels
// =====================================================================
__global__ void cvt_f32_f16_kernel(const float* __restrict__ src,
                                   h16* __restrict__ dst, int n) {
  int stride = gridDim.x * blockDim.x;
  for (int i = blockIdx.x * blockDim.x + threadIdx.x; i < n; i += stride)
    dst[i] = (h16)src[i];
}

// WkT[e*D + d] = Wk[d*D + e]   (so qk = q @ Wk becomes an NT GEMM)
__global__ void transpose_cvt_kernel(const float* __restrict__ W,
                                     h16* __restrict__ WT, int n /*D*D*/) {
  int stride = gridDim.x * blockDim.x;
  for (int i = blockIdx.x * blockDim.x + threadIdx.x; i < n; i += stride) {
    int r = i >> 8;          // d
    int c = i & 255;         // e
    WT[(size_t)c * D_ + r] = (h16)W[i];
  }
}

// biasq[k*D+d] = normalize(slot_queries)[k][d] + bus[k*D+d]
__global__ void slotq_bias_kernel(const float* __restrict__ slotq,
                                  const float* __restrict__ bus,
                                  float* __restrict__ out) {
  __shared__ float red[8];
  __shared__ float invn;
  int k = blockIdx.x;          // 0..K-1
  int d = threadIdx.x;         // 0..255
  float v = slotq[k * D_ + d];
  float ss = v * v;
  for (int o = 16; o; o >>= 1) ss += __shfl_xor(ss, o, 32);
  if ((threadIdx.x & 31) == 0) red[threadIdx.x >> 5] = ss;
  __syncthreads();
  if (threadIdx.x == 0) {
    float t = 0.f;
    for (int i = 0; i < 8; ++i) t += red[i];
    invn = 1.0f / fmaxf(sqrtf(t), 1e-12f);
  }
  __syncthreads();
  out[k * D_ + d] = v * invn + bus[k * D_ + d];
}

// gather rows by index and L2-normalize; one wave per row (8 rows/block)
__global__ void gather_norm_kernel(const float* __restrict__ table,
                                   const int* __restrict__ idx, int nrows,
                                   float* __restrict__ outf,
                                   h16* __restrict__ outh) {
  int row = blockIdx.x * 8 + (threadIdx.x >> 5);
  int lane = threadIdx.x & 31;
  if (row >= nrows) return;
  int bi = idx[row]; if (bi < 0) bi = 0;
  const float* src = table + (size_t)bi * D_;
  float v[8]; float ss = 0.f;
#pragma unroll
  for (int j = 0; j < 8; ++j) { v[j] = src[lane * 8 + j]; ss += v[j] * v[j]; }
  for (int o = 16; o; o >>= 1) ss += __shfl_xor(ss, o, 32);
  float inv = 1.0f / fmaxf(sqrtf(ss), 1e-12f);
  float* dstf = outf + (size_t)row * D_ + lane * 8;
  h16*   dsth = outh + (size_t)row * D_ + lane * 8;
#pragma unroll
  for (int j = 0; j < 8; ++j) { float x = v[j] * inv; dstf[j] = x; dsth[j] = (h16)x; }
}

// normalize rows of z [MROWS, D] in place; one wave per row
__global__ void row_norm_kernel(float* __restrict__ z, int nrows) {
  int row = blockIdx.x * 8 + (threadIdx.x >> 5);
  int lane = threadIdx.x & 31;
  if (row >= nrows) return;
  float* p = z + (size_t)row * D_ + lane * 8;
  float v[8]; float ss = 0.f;
#pragma unroll
  for (int j = 0; j < 8; ++j) { v[j] = p[j]; ss += v[j] * v[j]; }
  for (int o = 16; o; o >>= 1) ss += __shfl_xor(ss, o, 32);
  float inv = 1.0f / fmaxf(sqrtf(ss), 1e-12f);
#pragma unroll
  for (int j = 0; j < 8; ++j) p[j] = v[j] * inv;
}

// encoder feature: feat[m] = [sv, q, sv*q, |sv-q|] (f16, width 4*D)
__global__ void feat_enc_kernel(const float* __restrict__ sv,
                                const float* __restrict__ q,
                                h16* __restrict__ feat) {
  const int total = MROWS_ * D_;
  int stride = gridDim.x * blockDim.x;
  for (int i = blockIdx.x * blockDim.x + threadIdx.x; i < total; i += stride) {
    int m = i >> 8, d = i & 255;
    float s = sv[i], qq = q[i];
    h16* f = feat + (size_t)m * (4 * D_);
    f[d]            = (h16)s;
    f[D_ + d]       = (h16)qq;
    f[2 * D_ + d]   = (h16)(s * qq);
    f[3 * D_ + d]   = (h16)fabsf(s - qq);
  }
}

// denoiser feature: f[m] = [z, u, z*u, |z-u|, time_emb(step)] (f16, width 1152)
__global__ void feat_dn_kernel(const float* __restrict__ z,
                               const float* __restrict__ uv,
                               h16* __restrict__ f, int step) {
  const int total = MROWS_ * IN_DIM_;
  int stride = gridDim.x * blockDim.x;
  for (int i = blockIdx.x * blockDim.x + threadIdx.x; i < total; i += stride) {
    int m = i / IN_DIM_;
    int c = i - m * IN_DIM_;
    float val;
    if (c < 4 * D_) {
      int sect = c >> 8, d = c & 255;
      float zz = z[(size_t)m * D_ + d];
      float uu = uv[(size_t)(m >> 2) * D_ + d];
      val = (sect == 0) ? zz : (sect == 1) ? uu
          : (sect == 2) ? zz * uu : fabsf(zz - uu);
    } else {
      int j  = c - 4 * D_;              // 0..127
      int jj = j & 63;
      float fr  = __expf(-9.210340371976184f * (float)jj * (1.0f / 64.0f));
      float arg = (float)step * fr;
      val = (j < 64) ? __cosf(arg) : __sinf(arg);
    }
    f[i] = (h16)val;
  }
}

// =====================================================================
// Fused per-row attention: normalize observed bundles into LDS,
// scores = qk . obs, masked softmax, ao = attn @ obs  (K=4, L=50)
// =====================================================================
__global__ void __launch_bounds__(256)
attn_kernel(const float* __restrict__ bund, const int* __restrict__ obs_idx,
            const int* __restrict__ obs_mask, const h16* __restrict__ qk,
            h16* __restrict__ ao, float* __restrict__ attnsum) {
  __shared__ float obs[L_][D_];     // 51.2 KB (WGP has 320 KB)
  __shared__ float sc[K_][64];
  int b = blockIdx.x;
  int wave = threadIdx.x >> 5, lane = threadIdx.x & 31;

  // phase 1: gather + normalize 50 bundle rows into LDS (wave per row)
  for (int l = wave; l < L_; l += 8) {
    int bi = obs_idx[b * L_ + l]; if (bi < 0) bi = 0;
    const float* src = bund + (size_t)bi * D_;
    float v[8]; float ss = 0.f;
#pragma unroll
    for (int j = 0; j < 8; ++j) { v[j] = src[lane * 8 + j]; ss += v[j] * v[j]; }
    for (int o = 16; o; o >>= 1) ss += __shfl_xor(ss, o, 32);
    float inv = 1.0f / fmaxf(sqrtf(ss), 1e-12f);
#pragma unroll
    for (int j = 0; j < 8; ++j) obs[l][lane * 8 + j] = v[j] * inv;
  }
  __syncthreads();

  // phase 2: scores (bk is constant over l -> softmax-invariant, omitted)
  for (int p = wave; p < K_ * L_; p += 8) {
    int k = p / L_, l = p - k * L_;
    const h16* qp = qk + ((size_t)b * K_ + k) * D_;
    float dacc = 0.f;
#pragma unroll
    for (int j = 0; j < 8; ++j)
      dacc += (float)qp[lane * 8 + j] * obs[l][lane * 8 + j];
    for (int o = 16; o; o >>= 1) dacc += __shfl_xor(dacc, o, 32);
    if (lane == 0) {
      bool valid = obs_mask[b * L_ + l] > 0;
      sc[k][l] = valid ? dacc * 0.0625f /* 1/sqrt(D) */ : -1e30f;
    }
  }
  __syncthreads();

  // phase 3: masked softmax over l, one wave per slot k
  if (wave < K_) {
    int k = wave;
    float s0 = (lane < L_) ? sc[k][lane] : -1e30f;
    float s1 = (lane + 32 < L_) ? sc[k][lane + 32] : -1e30f;
    float mx = fmaxf(s0, s1);
    for (int o = 16; o; o >>= 1) mx = fmaxf(mx, __shfl_xor(mx, o, 32));
    float e0 = (lane < L_) ? __expf(s0 - mx) : 0.f;
    float e1 = (lane + 32 < L_) ? __expf(s1 - mx) : 0.f;
    float sm = e0 + e1;
    for (int o = 16; o; o >>= 1) sm += __shfl_xor(sm, o, 32);
    float inv = 1.0f / sm;
    bool v0 = (lane < L_) && (obs_mask[b * L_ + lane] > 0);
    bool v1 = (lane + 32 < L_) && (obs_mask[b * L_ + lane + 32] > 0);
    float a0 = v0 ? e0 * inv : 0.f;
    float a1 = v1 ? e1 * inv : 0.f;
    if (lane < L_) sc[k][lane] = a0;
    if (lane + 32 < L_) sc[k][lane + 32] = a1;
    float asum = a0 + a1;
    for (int o = 16; o; o >>= 1) asum += __shfl_xor(asum, o, 32);
    if (lane == 0) attnsum[b * K_ + k] = asum;
  }
  __syncthreads();

  // phase 4: ao[k] = sum_l attn[k][l] * obs[l]   (2 waves per slot)
  {
    int k = wave >> 1;
    int dbase = (wave & 1) * 128 + lane * 4;
    float acc[4] = {0.f, 0.f, 0.f, 0.f};
    for (int l = 0; l < L_; ++l) {
      float a = sc[k][l];
      const float* o = &obs[l][dbase];
      acc[0] += a * o[0]; acc[1] += a * o[1];
      acc[2] += a * o[2]; acc[3] += a * o[3];
    }
    h16* dst = ao + ((size_t)b * K_ + k) * D_ + dbase;
#pragma unroll
    for (int j = 0; j < 4; ++j) dst[j] = (h16)acc[j];
  }
}

// =====================================================================
// NT GEMM with WMMA + async W staging.
//   C[m,n] = sum_k A[m,k] * W[n,k] (+bias / rowscale*rsvec / relu)
// Block: 8 waves, 256 rows x 64 cols. Wave tile 32x64 (8 WMMA / K-step).
// The 64x32 W tile is identical for all 8 waves -> staged once per block
// into LDS with global_load_async_to_lds_b128 (ASYNCcnt), double-buffered;
// split-wait s_wait_asynccnt 0x1 keeps the next DMA in flight during WMMA.
// Fragment layout per CDNA5 ISA 16-bit A/B 16x32:
//   lane L holds row L%16; K-octets at k0 + 8*(L>=16) and +16.
// =====================================================================
__device__ __forceinline__ void async_w_load(h16* dst_lds, const h16* src) {
  unsigned int loff = (unsigned int)(size_t)dst_lds;   // LDS byte offset
  unsigned long long ga = (unsigned long long)(size_t)src;
  asm volatile("global_load_async_to_lds_b128 %0, %1, off"
               :: "v"(loff), "v"(ga) : "memory");
}

__device__ __forceinline__ v16h load_frag(const h16* __restrict__ base,
                                          int ld, int row, int k0, int lane) {
  const h16* p = base + (size_t)(row + (lane & 15)) * (size_t)ld
                      + (size_t)(k0 + ((lane & 16) >> 1));
  v8h lo = *(const v8h*)p;          // K .. K+7
  v8h hi = *(const v8h*)(p + 16);   // K+16 .. K+23
  return __builtin_shufflevector(lo, hi, 0, 1, 2, 3, 4, 5, 6, 7,
                                 8, 9, 10, 11, 12, 13, 14, 15);
}

__global__ void __launch_bounds__(256)
gemm_nt_kernel(const h16* __restrict__ A, const h16* __restrict__ W,
               int M, int N, int Kd,
               const float* __restrict__ bias,      // may be null
               const float* __restrict__ rowscale,  // may be null
               const float* __restrict__ rsvec,     // used with rowscale
               int relu,
               float* __restrict__ outF,            // may be null
               h16* __restrict__ outH) {            // may be null
  __shared__ h16 wt[2][64 * 32];                    // 2 x 4KB W tiles
  int tiles_n = N >> 6;
  int bm = blockIdx.x / tiles_n;
  int bn = blockIdx.x - bm * tiles_n;
  int tid = threadIdx.x;
  int wave = tid >> 5, lane = tid & 31;
  int m0 = bm * 256 + wave * 32;
  int n0 = bn * 64;

  // per-thread slice of the 64x32 W tile: row tid/4, 8-half chunk tid%4
  const h16* wsrc = W + (size_t)(n0 + (tid >> 2)) * (size_t)Kd + ((tid & 3) << 3);
  h16* wdst0 = &wt[0][tid << 3];
  h16* wdst1 = &wt[1][tid << 3];

  async_w_load(wdst0, wsrc);                        // prologue: tile kk=0

  v8f acc[8] = {};
  int buf = 0;
  for (int kk = 0; kk < Kd; kk += 32) {
    bool more = (kk + 32) < Kd;
    if (more) async_w_load(buf ? wdst0 : wdst1, wsrc + kk + 32);

    // A fragments from global; issued before the async wait to overlap
    v16h a0 = load_frag(A, Kd, m0,      kk, lane);
    v16h a1 = load_frag(A, Kd, m0 + 16, kk, lane);

    if (more) asm volatile("s_wait_asynccnt 0x1" ::: "memory");
    else      asm volatile("s_wait_asynccnt 0x0" ::: "memory");
    __syncthreads();

    const h16* wb = &wt[buf][0];
#pragma unroll
    for (int j = 0; j < 4; ++j) {
      const h16* wp = wb + (16 * j + (lane & 15)) * 32 + ((lane & 16) >> 1);
      v8h lo = *(const v8h*)wp;
      v8h hi = *(const v8h*)(wp + 16);
      v16h wf = __builtin_shufflevector(lo, hi, 0, 1, 2, 3, 4, 5, 6, 7,
                                        8, 9, 10, 11, 12, 13, 14, 15);
      acc[j]     = __builtin_amdgcn_wmma_f32_16x16x32_f16(
                       false, a0, false, wf, (short)0, acc[j],     false, false);
      acc[4 + j] = __builtin_amdgcn_wmma_f32_16x16x32_f16(
                       false, a1, false, wf, (short)0, acc[4 + j], false, false);
    }
    __syncthreads();        // all reads of wt[buf] done before it is refilled
    buf ^= 1;
  }

  // Epilogue. C layout: VGPR r -> M = m0 + r + 8*(lane>=16), N = n0+16j+(lane&15)
  int mb = m0 + ((lane & 16) >> 1);
  int nl = lane & 15;
#pragma unroll
  for (int j = 0; j < 4; ++j) {
    int n = n0 + 16 * j + nl;
    float bn_ = bias ? bias[n] : 0.f;
    float rv  = rowscale ? rsvec[n] : 0.f;
#pragma unroll
    for (int r = 0; r < 8; ++r) {
      int m = mb + r;
      float v = acc[j][r] + bn_;
      if (rowscale) v += rowscale[m] * rv;
      if (relu) v = fmaxf(v, 0.f);
      size_t off = (size_t)m * (size_t)N + (size_t)n;
      if (outF) outF[off] = v;
      if (outH) outH[off] = (h16)v;

      int m2 = m + 16;
      float v2 = acc[4 + j][r] + bn_;
      if (rowscale) v2 += rowscale[m2] * rv;
      if (relu) v2 = fmaxf(v2, 0.f);
      size_t off2 = (size_t)m2 * (size_t)N + (size_t)n;
      if (outF) outF[off2] = v2;
      if (outH) outH[off2] = (h16)v2;
    }
  }
}

// =====================================================================
// Host-side orchestration
// =====================================================================
static inline void* ws_take(char* base, size_t& off, size_t bytes) {
  void* p = base + off;
  off = (off + bytes + 255) & ~(size_t)255;
  return p;
}

extern "C" void kernel_launch(void* const* d_in, const int* in_sizes, int n_in,
                              void* d_out, int out_size, void* d_ws, size_t ws_size,
                              hipStream_t stream) {
  (void)in_sizes; (void)n_in; (void)out_size; (void)ws_size;
  const float* user_emb   = (const float*)d_in[0];
  const float* bundle_emb = (const float*)d_in[1];
  const float* slotq      = (const float*)d_in[2];
  const float* Wk  = (const float*)d_in[3];
  const float* bk  = (const float*)d_in[4];  (void)bk; // softmax-invariant
  const float* Wv  = (const float*)d_in[5];
  const float* bv  = (const float*)d_in[6];
  const float* Wus = (const float*)d_in[7];
  const float* bus = (const float*)d_in[8];
  const float* Wf1 = (const float*)d_in[9];
  const float* bf1 = (const float*)d_in[10];
  const float* Wf2 = (const float*)d_in[11];
  const float* bf2 = (const float*)d_in[12];
  const float* Wd1 = (const float*)d_in[13];
  const float* bd1 = (const float*)d_in[14];
  const float* Wd2 = (const float*)d_in[15];
  const float* bd2 = (const float*)d_in[16];
  const float* Wd3 = (const float*)d_in[17];
  const float* bd3 = (const float*)d_in[18];
  const int* uidx     = (const int*)d_in[19];
  const int* obs_idx  = (const int*)d_in[20];
  const int* obs_mask = (const int*)d_in[21];

  char* ws = (char*)d_ws;
  size_t off = 0;
  // f16 weight copies
  h16* wus_h = (h16*)ws_take(ws, off, (size_t)HID_ * D_ * 2);
  h16* wkT_h = (h16*)ws_take(ws, off, (size_t)D_ * D_ * 2);
  h16* wv_h  = (h16*)ws_take(ws, off, (size_t)D_ * D_ * 2);
  h16* wf1_h = (h16*)ws_take(ws, off, (size_t)HID_ * 4 * D_ * 2);
  h16* wf2_h = (h16*)ws_take(ws, off, (size_t)D_ * HID_ * 2);
  h16* wd1_h = (h16*)ws_take(ws, off, (size_t)HID_ * IN_DIM_ * 2);
  h16* wd2_h = (h16*)ws_take(ws, off, (size_t)HID_ * HID_ * 2);
  h16* wd3_h = (h16*)ws_take(ws, off, (size_t)D_ * HID_ * 2);
  // activations
  float* uv_f    = (float*)ws_take(ws, off, (size_t)B_ * D_ * 4);
  h16*   uv_h    = (h16*)ws_take(ws, off, (size_t)B_ * D_ * 2);
  float* q_f     = (float*)ws_take(ws, off, (size_t)MROWS_ * D_ * 4);
  h16*   q_h     = (h16*)ws_take(ws, off, (size_t)MROWS_ * D_ * 2);
  h16*   qk_h    = (h16*)ws_take(ws, off, (size_t)MROWS_ * D_ * 2);
  h16*   ao_h    = (h16*)ws_take(ws, off, (size_t)MROWS_ * D_ * 2);
  float* sv_f    = (float*)ws_take(ws, off, (size_t)MROWS_ * D_ * 4);
  float* z_f     = (float*)ws_take(ws, off, (size_t)MROWS_ * D_ * 4);
  float* biasq   = (float*)ws_take(ws, off, (size_t)K_ * D_ * 4);
  float* attnsum = (float*)ws_take(ws, off, (size_t)MROWS_ * 4);
  h16*   feat_h  = (h16*)ws_take(ws, off, (size_t)MROWS_ * IN_DIM_ * 2); // covers 4D & IN_DIM
  h16*   h1_h    = (h16*)ws_take(ws, off, (size_t)MROWS_ * HID_ * 2);
  h16*   h2_h    = (h16*)ws_take(ws, off, (size_t)MROWS_ * HID_ * 2);

  dim3 blk(256);

  // ---- weight conversion (f32 -> f16, Wk transposed) ----
  cvt_f32_f16_kernel<<<512, blk, 0, stream>>>(Wus, wus_h, HID_ * D_);
  transpose_cvt_kernel<<<256, blk, 0, stream>>>(Wk, wkT_h, D_ * D_);
  cvt_f32_f16_kernel<<<256, blk, 0, stream>>>(Wv, wv_h, D_ * D_);
  cvt_f32_f16_kernel<<<1024, blk, 0, stream>>>(Wf1, wf1_h, HID_ * 4 * D_);
  cvt_f32_f16_kernel<<<512, blk, 0, stream>>>(Wf2, wf2_h, D_ * HID_);
  cvt_f32_f16_kernel<<<1024, blk, 0, stream>>>(Wd1, wd1_h, HID_ * IN_DIM_);
  cvt_f32_f16_kernel<<<1024, blk, 0, stream>>>(Wd2, wd2_h, HID_ * HID_);
  cvt_f32_f16_kernel<<<512, blk, 0, stream>>>(Wd3, wd3_h, D_ * HID_);

  // ---- uv = normalize(user_emb[uidx]); biasq = normalize(slotq)+bus ----
  gather_norm_kernel<<<B_ / 8, blk, 0, stream>>>(user_emb, uidx, B_, uv_f, uv_h);
  slotq_bias_kernel<<<K_, blk, 0, stream>>>(slotq, bus, biasq);

  // ---- q = uv @ Wus^T + (biasq broadcast)  [2048,1024] == [8192,256] ----
  gemm_nt_kernel<<<(B_ / 256) * (HID_ / 64), blk, 0, stream>>>(
      uv_h, wus_h, B_, HID_, D_, biasq, nullptr, nullptr, 0, q_f, q_h);

  // ---- qk = q @ Wk  (NT against Wk^T)  [8192,256] ----
  gemm_nt_kernel<<<(MROWS_ / 256) * (D_ / 64), blk, 0, stream>>>(
      q_h, wkT_h, MROWS_, D_, D_, nullptr, nullptr, nullptr, 0, nullptr, qk_h);

  // ---- fused attention: scores/softmax/ao + attnsum ----
  attn_kernel<<<B_, blk, 0, stream>>>(bundle_emb, obs_idx, obs_mask,
                                      qk_h, ao_h, attnsum);

  // ---- sv = ao @ Wv^T + attnsum * bv  [8192,256] ----
  gemm_nt_kernel<<<(MROWS_ / 256) * (D_ / 64), blk, 0, stream>>>(
      ao_h, wv_h, MROWS_, D_, D_, nullptr, attnsum, bv, 0, sv_f, nullptr);

  // ---- feat = [sv, q, sv*q, |sv-q|]  f16 [8192,1024] ----
  feat_enc_kernel<<<2048, blk, 0, stream>>>(sv_f, q_f, feat_h);

  // ---- h = relu(feat @ Wf1^T + bf1) ; slots = h @ Wf2^T + bf2 ----
  gemm_nt_kernel<<<(MROWS_ / 256) * (HID_ / 64), blk, 0, stream>>>(
      feat_h, wf1_h, MROWS_, HID_, 4 * D_, bf1, nullptr, nullptr, 1, nullptr, h1_h);
  gemm_nt_kernel<<<(MROWS_ / 256) * (D_ / 64), blk, 0, stream>>>(
      h1_h, wf2_h, MROWS_, D_, HID_, bf2, nullptr, nullptr, 0, z_f, nullptr);

  // ---- z = normalize(slots) per (b,k) row ----
  row_norm_kernel<<<MROWS_ / 8, blk, 0, stream>>>(z_f, MROWS_);

  // ---- denoiser: 10 reverse steps ----
  for (int step = NSTEPS_ - 1; step >= 0; --step) {
    feat_dn_kernel<<<4096, blk, 0, stream>>>(z_f, uv_f, feat_h, step);
    gemm_nt_kernel<<<(MROWS_ / 256) * (HID_ / 64), blk, 0, stream>>>(
        feat_h, wd1_h, MROWS_, HID_, IN_DIM_, bd1, nullptr, nullptr, 1,
        nullptr, h1_h);
    gemm_nt_kernel<<<(MROWS_ / 256) * (HID_ / 64), blk, 0, stream>>>(
        h1_h, wd2_h, MROWS_, HID_, HID_, bd2, nullptr, nullptr, 1,
        nullptr, h2_h);
    float* zout = (step == 0) ? (float*)d_out : z_f;
    gemm_nt_kernel<<<(MROWS_ / 256) * (D_ / 64), blk, 0, stream>>>(
        h2_h, wd3_h, MROWS_, D_, HID_, bd3, nullptr, nullptr, 0,
        zout, nullptr);
  }
}
